// GATLayer_68101001445815
// MI455X (gfx1250) — matrix operand order repaired and checked by reading, compiled, not run on hardware
//
#include <hip/hip_runtime.h>
#include <hip/hip_bf16.h>
#include <math.h>

typedef __attribute__((ext_vector_type(16))) _Float16 v16h;
typedef __attribute__((ext_vector_type(8)))  float    v8f;

#define IN_DIM 128
#define NHEAD  8
#define HD     16   // per-head out dim

// ---------------------------------------------------------------------------
// GEMM: z[N, 128] = h[N,128] @ W, where W[k][head*16+d] = Wfc[head][k][d].
// One wave -> one 16(M) x 16(N=one head) tile, K=128 in 4 WMMA steps of 32.
// blockDim = 256 (8 waves); wave w == head w; blockIdx.x == row tile.
// ---------------------------------------------------------------------------
__global__ __launch_bounds__(256) void gat1250_gemm_wmma(
    const float* __restrict__ h, const float* __restrict__ Wfc,
    float* __restrict__ z, int N)
{
    const int head = threadIdx.x >> 5;          // 0..7
    const int lane = threadIdx.x & 31;
    const int hl   = lane & 15;
    const bool hi  = lane >= 16;
    const int m0   = blockIdx.x * 16;

    int rowA = m0 + hl;
    if (rowA >= N) rowA = N - 1;                // clamp (keeps EXEC all-ones)
    const float* __restrict__ hrow = h + (size_t)rowA * IN_DIM;
    const float* __restrict__ wh   = Wfc + (size_t)head * (IN_DIM * HD);

    v8f c = {};
    #pragma unroll
    for (int k0 = 0; k0 < IN_DIM; k0 += 32) {
        // ---- A tile (16x32 f16): lanes0-15 K in {0-7,16-23}, lanes16-31 +8
        const int offA = k0 + (hi ? 8 : 0);
        float fa[16];
        *(float4*)(fa + 0)  = *(const float4*)(hrow + offA + 0);
        *(float4*)(fa + 4)  = *(const float4*)(hrow + offA + 4);
        *(float4*)(fa + 8)  = *(const float4*)(hrow + offA + 16);
        *(float4*)(fa + 12) = *(const float4*)(hrow + offA + 20);
        v16h a;
        #pragma unroll
        for (int j = 0; j < 16; ++j) a[j] = (_Float16)fa[j];

        // ---- B tile (32x16 f16): lane holds col hl; lanes0-15 K in {0-7,16-23}, hi +8
        const int offB = k0 + (hi ? 8 : 0);
        v16h b;
        #pragma unroll
        for (int j = 0; j < 16; ++j)
            b[j] = (_Float16)wh[(size_t)(offB + j + (j & 8)) * HD + hl];

        c = __builtin_amdgcn_wmma_f32_16x16x32_f16(
                /*neg_a=*/false, a, /*neg_b=*/false, b,
                /*c_mod=*/(short)0, c, /*reuse_a=*/false, /*reuse_b=*/false);
    }

    // D layout: VGPR r -> row m0 + (hi?8:0) + r, col = head*16 + hl
    const int rbase = m0 + (hi ? 8 : 0);
    #pragma unroll
    for (int r = 0; r < 8; ++r) {
        const int row = rbase + r;
        if (row < N) z[(size_t)row * IN_DIM + head * HD + hl] = c[r];
    }
}

// ---------------------------------------------------------------------------
// Workspace init: m = -FLT_MAX, denom = 0, num = 0
// ---------------------------------------------------------------------------
__global__ void gat1250_init(float* __restrict__ m, float* __restrict__ denom,
                             float* __restrict__ num, int total)
{
    const int t = blockIdx.x * blockDim.x + threadIdx.x;
    if (t < total) {
        m[t]     = -3.402823466e38f;
        denom[t] = 0.0f;
        num[t]   = 0.0f;
    }
}

// Sign-aware float atomic max built on native int atomics (no CAS loop).
__device__ __forceinline__ void atomicMaxF32(float* addr, float v)
{
    if (v >= 0.0f) atomicMax((int*)addr, __float_as_int(v));
    else           atomicMin((unsigned int*)addr, __float_as_uint(v));
}

// ---------------------------------------------------------------------------
// Edge pass 1: per-channel segment max of e = z[src]*z[dst] over dst.
// One wave per edge, 4 channels per lane (float4).
// ---------------------------------------------------------------------------
__global__ __launch_bounds__(256) void gat1250_edge_max(
    const float* __restrict__ z, const int* __restrict__ src,
    const int* __restrict__ dst, float* __restrict__ m, int E)
{
    const int t    = blockIdx.x * blockDim.x + threadIdx.x;
    const int e    = t >> 5;
    const int lane = t & 31;
    if (e >= E) return;
    const int s = src[e], d = dst[e];
    const float4 a = ((const float4*)(z + (size_t)s * IN_DIM))[lane];
    const float4 b = ((const float4*)(z + (size_t)d * IN_DIM))[lane];
    float* mp = m + (size_t)d * IN_DIM + lane * 4;
    atomicMaxF32(mp + 0, a.x * b.x);
    atomicMaxF32(mp + 1, a.y * b.y);
    atomicMaxF32(mp + 2, a.z * b.z);
    atomicMaxF32(mp + 3, a.w * b.w);
}

// ---------------------------------------------------------------------------
// Edge pass 2: ex = exp(e - m[dst]); denom[dst] += ex; num[dst] += ex*z[src].
// (agg = num/denom afterwards: Σ alpha*zs == Σ ex*zs / Σ ex)
// ---------------------------------------------------------------------------
__global__ __launch_bounds__(256) void gat1250_edge_expsum(
    const float* __restrict__ z, const int* __restrict__ src,
    const int* __restrict__ dst, const float* __restrict__ m,
    float* __restrict__ denom, float* __restrict__ num, int E)
{
    const int t    = blockIdx.x * blockDim.x + threadIdx.x;
    const int e    = t >> 5;
    const int lane = t & 31;
    if (e >= E) return;
    const int s = src[e], d = dst[e];
    const float4 a  = ((const float4*)(z + (size_t)s * IN_DIM))[lane];
    const float4 b  = ((const float4*)(z + (size_t)d * IN_DIM))[lane];
    const float4 mm = ((const float4*)(m + (size_t)d * IN_DIM))[lane];
    float4 ex;
    ex.x = __expf(a.x * b.x - mm.x);
    ex.y = __expf(a.y * b.y - mm.y);
    ex.z = __expf(a.z * b.z - mm.z);
    ex.w = __expf(a.w * b.w - mm.w);
    const size_t base = (size_t)d * IN_DIM + lane * 4;
    atomicAdd(denom + base + 0, ex.x);
    atomicAdd(denom + base + 1, ex.y);
    atomicAdd(denom + base + 2, ex.z);
    atomicAdd(denom + base + 3, ex.w);
    atomicAdd(num + base + 0, ex.x * a.x);
    atomicAdd(num + base + 1, ex.y * a.y);
    atomicAdd(num + base + 2, ex.z * a.z);
    atomicAdd(num + base + 3, ex.w * a.w);
}

// ---------------------------------------------------------------------------
// Finalize: out = h + elu((num/denom) * snorm_n)
// ---------------------------------------------------------------------------
__global__ __launch_bounds__(256) void gat1250_finalize(
    const float* __restrict__ h, const float* __restrict__ snorm,
    const float* __restrict__ denom, const float* __restrict__ num,
    float* __restrict__ out, int total)
{
    const int t = blockIdx.x * blockDim.x + threadIdx.x;
    if (t >= total) return;
    const int n = t >> 7;               // / IN_DIM
    const float den = denom[t];
    const float agg = den > 0.0f ? num[t] / den : 0.0f;   // empty segment -> 0
    const float v = agg * snorm[n];
    const float o = v > 0.0f ? v : expm1f(v);
    out[t] = h[t] + o;
}

// ---------------------------------------------------------------------------
extern "C" void kernel_launch(void* const* d_in, const int* in_sizes, int n_in,
                              void* d_out, int out_size, void* d_ws, size_t ws_size,
                              hipStream_t stream)
{
    const float* h     = (const float*)d_in[0];
    const float* snorm = (const float*)d_in[1];
    const float* Wfc   = (const float*)d_in[2];
    const int*   src   = (const int*)d_in[3];
    const int*   dst   = (const int*)d_in[4];
    float*       out   = (float*)d_out;

    const int N = in_sizes[1];          // snorm_n has N elements
    const int E = in_sizes[3];
    const int total = N * IN_DIM;

    float* z     = (float*)d_ws;        // N*128
    float* m     = z + (size_t)total;   // N*128
    float* denom = m + (size_t)total;   // N*128
    float* num   = denom + (size_t)total; // N*128

    // 1) projection GEMM (WMMA): one row-tile per block, one head per wave
    const int mtiles = (N + 15) / 16;
    hipLaunchKernelGGL(gat1250_gemm_wmma, dim3(mtiles), dim3(256), 0, stream,
                       h, Wfc, z, N);

    // 2) init segment buffers
    hipLaunchKernelGGL(gat1250_init, dim3((total + 255) / 256), dim3(256), 0, stream,
                       m, denom, num, total);

    // 3) segment max over edges
    const long edge_threads = (long)E * 32;
    hipLaunchKernelGGL(gat1250_edge_max,
                       dim3((unsigned)((edge_threads + 255) / 256)), dim3(256), 0, stream,
                       z, src, dst, m, E);

    // 4) exp + segment sums (denominator and numerator in one pass)
    hipLaunchKernelGGL(gat1250_edge_expsum,
                       dim3((unsigned)((edge_threads + 255) / 256)), dim3(256), 0, stream,
                       z, src, dst, m, denom, num, E);

    // 5) normalize, graph-norm, ELU, residual
    hipLaunchKernelGGL(gat1250_finalize, dim3((total + 255) / 256), dim3(256), 0, stream,
                       h, snorm, denom, num, out, total);
}
